// SetAttentionAggregatorPro_71090298683768
// MI455X (gfx1250) — matrix operand order repaired and checked by reading, not compiled, not run on hardware
//
#include <hip/hip_runtime.h>
#include <hip/hip_bf16.h>

typedef __attribute__((ext_vector_type(16))) __bf16 v16bf;
typedef __attribute__((ext_vector_type(8)))  __bf16 v8bf;
typedef __attribute__((ext_vector_type(8)))  float  v8f;

namespace {
constexpr int kB = 8, kN = 8192, kD = 256, kH = 8, kDH = 32, kM = 64;
constexpr int kRows = kB * kN;                  // 65536
constexpr float kDN   = 0.42044820762685725f;   // DH^-0.25
constexpr float kEpsK = 1e-3f;
constexpr float kLnEps = 1e-5f;
}

__device__ __forceinline__ v8f vzero8() {
  v8f z;
#pragma unroll
  for (int i = 0; i < 8; ++i) z[i] = 0.f;
  return z;
}

// D = A(16x32 bf16) * B(32x16 bf16) + C (f32)
__device__ __forceinline__ v8f wmma_bf16(v16bf a, v16bf b, v8f c) {
  return __builtin_amdgcn_wmma_f32_16x16x32_bf16(
      /*neg_a=*/false, a, /*neg_b=*/false, b,
      /*c_mod=*/(short)0, c, /*reuse_a=*/false, /*reuse_b=*/false);
}

// A fragment from row-major [16 x K] tile; per-lane: rows lane&15,
// K chunks {hi*8.., 16+hi*8..} (ISA 16-bit A layout). 16B-aligned loads.
__device__ __forceinline__ v16bf load_a_frag(const __bf16* tile, int ld) {
  const int lane = threadIdx.x & 31;
  const __bf16* p = tile + (lane & 15) * ld + (lane >> 4) * 8;
  v8bf lo = *(const v8bf*)(p);
  v8bf hh = *(const v8bf*)(p + 16);
  v16bf a;
#pragma unroll
  for (int i = 0; i < 8; ++i) { a[i] = lo[i]; a[8 + i] = hh[i]; }
  return a;
}

// B fragment from an N-major array T[n*ld + k] (i.e. B transposed in memory);
// per-lane: col lane&15, K chunks {hi*8.., 16+hi*8..} (matches A layout).
__device__ __forceinline__ v16bf load_b_fragT(const __bf16* tileT, int ld) {
  const int lane = threadIdx.x & 31;
  const __bf16* p = tileT + (lane & 15) * ld + (lane >> 4) * 8;
  v8bf lo = *(const v8bf*)(p);
  v8bf hh = *(const v8bf*)(p + 16);
  v16bf b;
#pragma unroll
  for (int i = 0; i < 8; ++i) { b[i] = lo[i]; b[8 + i] = hh[i]; }
  return b;
}

// order-preserving float->uint encoding for atomicMax
__device__ __forceinline__ unsigned enc_f(float f) {
  unsigned u = __float_as_uint(f);
  return (u & 0x80000000u) ? ~u : (u | 0x80000000u);
}
__device__ __forceinline__ float dec_f(unsigned u) {
  return (u & 0x80000000u) ? __uint_as_float(u ^ 0x80000000u)
                           : __uint_as_float(~u);
}

// ---------------- kernel 1: repack weights (transpose + f32->bf16) ---------
__global__ void k_pack(const float* Wq, const float* Wk, const float* Wv,
                       const float* Wo, const float* proj,
                       __bf16* WqT, __bf16* WkT, __bf16* WvT, __bf16* WoT,
                       __bf16* projbf) {
  int idx = blockIdx.x * 256 + threadIdx.x;
  const int nW = 4 * kD * kD;
  if (idx < nW) {
    int mat = idx >> 16;
    int e = idx & 65535;
    int k = e >> 8, n = e & 255;
    const float* src = (mat == 0) ? Wq : (mat == 1) ? Wk : (mat == 2) ? Wv : Wo;
    __bf16* dst = (mat == 0) ? WqT : (mat == 1) ? WkT : (mat == 2) ? WvT : WoT;
    dst[n * kD + k] = (__bf16)src[k * kD + n];     // N-major for B fragments
  } else if (idx < nW + kM * kDH) {
    int e = idx - nW;
    projbf[e] = (__bf16)proj[e];                   // proj is already B^T layout
  }
}

// ---------------- kernel 2: LayerNorm(x) -> bf16 ---------------------------
__global__ void k_ln(const float* x, const float* g, const float* b, __bf16* xn) {
  int wave = threadIdx.x >> 5, lane = threadIdx.x & 31;
  int row = blockIdx.x * 8 + wave;
  const float* xr = x + (size_t)row * kD;
  float v[8], s = 0.f, s2 = 0.f;
#pragma unroll
  for (int i = 0; i < 8; ++i) {
    v[i] = xr[i * 32 + lane]; s += v[i]; s2 += v[i] * v[i];
  }
#pragma unroll
  for (int m = 1; m < 32; m <<= 1) {
    s += __shfl_xor(s, m, 32); s2 += __shfl_xor(s2, m, 32);
  }
  float mu = s * (1.f / kD);
  float var = s2 * (1.f / kD) - mu * mu;
  float rs = rsqrtf(var + kLnEps);
#pragma unroll
  for (int i = 0; i < 8; ++i) {
    int c = i * 32 + lane;
    xn[(size_t)row * kD + c] = (__bf16)((v[i] - mu) * rs * g[c] + b[c]);
  }
}

__global__ void k_zero(float* p, int n) {
  int i = blockIdx.x * 256 + threadIdx.x;
  if (i < n) p[i] = 0.f;
}

// ---------------- kernel 3: QKV GEMMs (16 rows x 256 cols per block) -------
__global__ void k_qkv(const __bf16* xn,
                      const __bf16* WqT, const __bf16* WkT, const __bf16* WvT,
                      const float* bq, const float* bk, const float* bv,
                      const unsigned char* mask,
                      __bf16* qb, __bf16* kb, __bf16* vb) {
  const int which = blockIdx.y;                 // 0=q 1=k 2=v
  const __bf16* WT  = (which == 0) ? WqT : (which == 1) ? WkT : WvT;
  const float* bias = (which == 0) ? bq  : (which == 1) ? bk  : bv;
  __bf16* out       = (which == 0) ? qb  : (which == 1) ? kb  : vb;
  const int row0 = blockIdx.x * 16;
  const int w = threadIdx.x >> 5, lane = threadIdx.x & 31;
  const int col0 = w * 32, coln = lane & 15, hi = lane >> 4;
  v8f acc0 = vzero8(), acc1 = vzero8();
#pragma unroll
  for (int kt = 0; kt < 8; ++kt) {
    v16bf a  = load_a_frag(xn + (size_t)row0 * kD + kt * 32, kD);
    v16bf b0 = load_b_fragT(WT + (size_t)col0 * kD + kt * 32, kD);
    v16bf b1 = load_b_fragT(WT + (size_t)(col0 + 16) * kD + kt * 32, kD);
    acc0 = wmma_bf16(a, b0, acc0);
    acc1 = wmma_bf16(a, b1, acc1);
  }
#pragma unroll
  for (int r = 0; r < 8; ++r) {
    int gr = row0 + r + 8 * hi;                 // flat (b*N + n)
    int c0 = col0 + coln, c1 = col0 + 16 + coln;
    float f0 = acc0[r] + bias[c0];
    float f1 = acc1[r] + bias[c1];
    if (which == 2) { float m = (float)mask[gr]; f0 *= m; f1 *= m; }
    out[(size_t)gr * kD + c0] = (__bf16)f0;
    out[(size_t)gr * kD + c1] = (__bf16)f1;
  }
}

// ------- kernel 4: fused kp = dn*relu(k@projT)+eps ; ksum ; ctx = kp^T v ---
__global__ void k_ctx(const __bf16* kb, const __bf16* vb, const __bf16* projbf,
                      float* ctx, float* ksum) {
  __shared__ __bf16 kpT[kM * 32];   // [m][n] 64x32
  __shared__ __bf16 vT[32 * 32];    // [d][n] 32x32
  const int split = blockIdx.x & 7;
  const int bh = blockIdx.x >> 3;
  const int h = bh & 7, b = bh >> 3;
  const int w = threadIdx.x >> 5, lane = threadIdx.x & 31;
  const int coln = lane & 15, hi = lane >> 4;
  const int ntile = w >> 2, mtile = w & 3;      // kp phase: 2x4 tiles
  const int mtile2 = w >> 1, dtile = w & 1;     // ctx phase: 4x2 tiles
  v8f ctxacc = vzero8();
  float kspart = 0.f;
  for (int c = 0; c < 32; ++c) {
    int nb = split * 1024 + c * 32;
    v16bf a  = load_a_frag(kb + (size_t)(b * kN + nb + ntile * 16) * kD + h * kDH, kD);
    v16bf pb = load_b_fragT(projbf + mtile * 16 * kDH, kDH);
    v8f d = wmma_bf16(a, pb, vzero8());
#pragma unroll
    for (int r = 0; r < 8; ++r) {
      float f = kDN * fmaxf(d[r], 0.f) + kEpsK;   // ksum over ALL n (ref semantics)
      kspart += f;
      kpT[(mtile * 16 + coln) * 32 + ntile * 16 + r + 8 * hi] = (__bf16)f;
    }
    { // stage v chunk transposed: vT[d][n]
      int t = threadIdx.x, nn = t & 31, dg = t >> 5;
#pragma unroll
      for (int j = 0; j < 4; ++j) {
        int dd = dg * 4 + j;
        vT[dd * 32 + nn] = vb[(size_t)(b * kN + nb + nn) * kD + h * kDH + dd];
      }
    }
    __syncthreads();
    v16bf a2 = load_a_frag(kpT + mtile2 * 16 * 32, 32);
    v16bf b2 = load_b_fragT(vT + dtile * 16 * 32, 32);
    ctxacc = wmma_bf16(a2, b2, ctxacc);
    __syncthreads();
  }
  float* cp = ctx + (size_t)bh * kM * kDH;
#pragma unroll
  for (int r = 0; r < 8; ++r)
    atomicAdd(&cp[(mtile2 * 16 + r + 8 * hi) * kDH + dtile * 16 + coln], ctxacc[r]);
  atomicAdd(&ksum[bh * kM + mtile * 16 + coln], kspart);
}

// ------- kernel 5: qp, d_inv, out = dinv * (qp @ ctx) ----------------------
__global__ void k_attn_out(const __bf16* qb, const __bf16* projbf,
                           const float* ctx, const float* ksum, __bf16* oa) {
  __shared__ __bf16 ctxT[32 * kM];       // [d][m]
  __shared__ float ksumS[kM];
  __shared__ __bf16 qpS[8][16 * kM];     // per-wave scaled qp, 16 x 64 row-major
  const int h = blockIdx.y;
  const int gr0 = blockIdx.x * 128;
  const int b = gr0 >> 13;
  const int bh = b * kH + h;
  const int w = threadIdx.x >> 5, lane = threadIdx.x & 31;
  const int coln = lane & 15, hi = lane >> 4;
  {
    int t = threadIdx.x;
    const float* cp = ctx + (size_t)bh * kM * kDH;
#pragma unroll
    for (int i = 0; i < 8; ++i) {
      int e = t * 8 + i, m = e >> 5, dd = e & 31;
      ctxT[dd * kM + m] = (__bf16)cp[m * kDH + dd];
    }
    if (t < kM) ksumS[t] = ksum[bh * kM + t];
  }
  __syncthreads();
  const int row0 = gr0 + w * 16;
  v16bf a = load_a_frag(qb + (size_t)row0 * kD + h * kDH, kD);
  float qpf[4][8];
#pragma unroll
  for (int t4 = 0; t4 < 4; ++t4) {
    v16bf pb = load_b_fragT(projbf + t4 * 16 * kDH, kDH);
    v8f d = wmma_bf16(a, pb, vzero8());
#pragma unroll
    for (int r = 0; r < 8; ++r) qpf[t4][r] = kDN * fmaxf(d[r], 0.f) + kEpsK;
  }
  float part[8];
#pragma unroll
  for (int r = 0; r < 8; ++r) {
    float p = 0.f;
#pragma unroll
    for (int t4 = 0; t4 < 4; ++t4) p += qpf[t4][r] * ksumS[t4 * 16 + coln];
    part[r] = p;
  }
#pragma unroll
  for (int m = 1; m <= 8; m <<= 1) {     // reduce over 16 cols (half-wave)
#pragma unroll
    for (int r = 0; r < 8; ++r) part[r] += __shfl_xor(part[r], m, 32);
  }
  __bf16* qs = &qpS[w][0];
#pragma unroll
  for (int t4 = 0; t4 < 4; ++t4) {
#pragma unroll
    for (int r = 0; r < 8; ++r)
      qs[(r + 8 * hi) * kM + t4 * 16 + coln] = (__bf16)(qpf[t4][r] / part[r]);
  }
  __syncthreads();
  v8f o0 = vzero8(), o1 = vzero8();
#pragma unroll
  for (int kt = 0; kt < 2; ++kt) {       // K = 64 over features
    v16bf a2  = load_a_frag(qs + kt * 32, kM);
    v16bf bb0 = load_b_fragT(ctxT + 0 * 16 * kM + kt * 32, kM);
    v16bf bb1 = load_b_fragT(ctxT + 1 * 16 * kM + kt * 32, kM);
    o0 = wmma_bf16(a2, bb0, o0);
    o1 = wmma_bf16(a2, bb1, o1);
  }
#pragma unroll
  for (int r = 0; r < 8; ++r) {
    int gr = row0 + r + 8 * hi;
    oa[(size_t)gr * kD + h * kDH + coln] = (__bf16)o0[r];
    oa[(size_t)gr * kD + h * kDH + 16 + coln] = (__bf16)o1[r];
  }
}

// ------- kernel 6: Wo GEMM + bias + residual + LN + masked pooling ---------
__global__ void k_wo_ln_pool(const __bf16* oa, const __bf16* WoT, const float* bo,
                             const float* x, const float* g2, const float* b2,
                             const unsigned char* mask,
                             float* sumpool, unsigned* maxenc) {
  __shared__ float rowsum[16], rowsq[16];
  __shared__ float colsum[kD];
  __shared__ unsigned colmax[kD];
  const int row0 = blockIdx.x * 16;
  const int b = row0 >> 13;
  const int w = threadIdx.x >> 5, lane = threadIdx.x & 31;
  const int col0 = w * 32, coln = lane & 15, hi = lane >> 4;
  if (threadIdx.x < 16) { rowsum[threadIdx.x] = 0.f; rowsq[threadIdx.x] = 0.f; }
  colsum[threadIdx.x] = 0.f;
  colmax[threadIdx.x] = 0u;   // encodes a very negative float
  __syncthreads();
  v8f acc0 = vzero8(), acc1 = vzero8();
#pragma unroll
  for (int kt = 0; kt < 8; ++kt) {
    v16bf a  = load_a_frag(oa + (size_t)row0 * kD + kt * 32, kD);
    v16bf b0 = load_b_fragT(WoT + (size_t)col0 * kD + kt * 32, kD);
    v16bf b1 = load_b_fragT(WoT + (size_t)(col0 + 16) * kD + kt * 32, kD);
    acc0 = wmma_bf16(a, b0, acc0);
    acc1 = wmma_bf16(a, b1, acc1);
  }
  const int c0 = col0 + coln, c1 = col0 + 16 + coln;
  float v0[8], v1[8];
#pragma unroll
  for (int r = 0; r < 8; ++r) {
    int gr = row0 + r + 8 * hi;
    v0[r] = acc0[r] + bo[c0] + x[(size_t)gr * kD + c0];
    v1[r] = acc1[r] + bo[c1] + x[(size_t)gr * kD + c1];
    atomicAdd(&rowsum[r + 8 * hi], v0[r] + v1[r]);
    atomicAdd(&rowsq[r + 8 * hi], v0[r] * v0[r] + v1[r] * v1[r]);
  }
  __syncthreads();
#pragma unroll
  for (int r = 0; r < 8; ++r) {
    int rl = r + 8 * hi, gr = row0 + rl;
    float mu = rowsum[rl] * (1.f / kD);
    float var = rowsq[rl] * (1.f / kD) - mu * mu;
    float rs = rsqrtf(var + kLnEps);
    float xr0 = (v0[r] - mu) * rs * g2[c0] + b2[c0];
    float xr1 = (v1[r] - mu) * rs * g2[c1] + b2[c1];
    if (mask[gr]) {
      atomicAdd(&colsum[c0], xr0);
      atomicAdd(&colsum[c1], xr1);
      atomicMax(&colmax[c0], enc_f(xr0));
      atomicMax(&colmax[c1], enc_f(xr1));
    }
  }
  __syncthreads();
  atomicAdd(&sumpool[b * kD + threadIdx.x], colsum[threadIdx.x]);
  atomicMax(&maxenc[b * kD + threadIdx.x], colmax[threadIdx.x]);
}

// ------- kernel 7: (max + sum/num_valid)/2 ---------------------------------
__global__ void k_finalize(const unsigned char* mask, const float* sumpool,
                           const unsigned* maxenc, float* out) {
  __shared__ int red[256];
  const int b = blockIdx.x;
  int c = 0;
  for (int n = threadIdx.x; n < kN; n += 256) c += (int)mask[b * kN + n];
  red[threadIdx.x] = c;
  __syncthreads();
  for (int s = 128; s > 0; s >>= 1) {
    if (threadIdx.x < s) red[threadIdx.x] += red[threadIdx.x + s];
    __syncthreads();
  }
  float nv = fmaxf((float)red[0], 1.f);
  float mean = sumpool[b * kD + threadIdx.x] / nv;
  float mx = dec_f(maxenc[b * kD + threadIdx.x]);
  out[b * kD + threadIdx.x] = 0.5f * (mx + mean);
}

extern "C" void kernel_launch(void* const* d_in, const int* in_sizes, int n_in,
                              void* d_out, int out_size, void* d_ws, size_t ws_size,
                              hipStream_t stream) {
  (void)in_sizes; (void)n_in; (void)out_size; (void)ws_size;
  const float* x    = (const float*)d_in[0];
  const unsigned char* mask = (const unsigned char*)d_in[1]; // jnp bool = 1 byte
  const float* g1   = (const float*)d_in[2];
  const float* b1   = (const float*)d_in[3];
  const float* Wq   = (const float*)d_in[4];
  const float* bq   = (const float*)d_in[5];
  const float* Wk   = (const float*)d_in[6];
  const float* bk   = (const float*)d_in[7];
  const float* Wv   = (const float*)d_in[8];
  const float* bv   = (const float*)d_in[9];
  const float* proj = (const float*)d_in[10];
  const float* Wo   = (const float*)d_in[11];
  const float* bo   = (const float*)d_in[12];
  const float* g2   = (const float*)d_in[13];
  const float* b2   = (const float*)d_in[14];
  float* out = (float*)d_out;

  char* ws = (char*)d_ws;
  const size_t SZA = (size_t)kRows * kD * sizeof(__bf16); // 32 MB
  __bf16* xn = (__bf16*)(ws);
  __bf16* qb = (__bf16*)(ws + SZA);
  __bf16* kb = (__bf16*)(ws + 2 * SZA);
  __bf16* vb = (__bf16*)(ws + 3 * SZA);
  __bf16* oa = (__bf16*)(ws + 4 * SZA);
  size_t off = 5 * SZA;
  __bf16* WqT = (__bf16*)(ws + off); off += (size_t)kD * kD * 2;
  __bf16* WkT = (__bf16*)(ws + off); off += (size_t)kD * kD * 2;
  __bf16* WvT = (__bf16*)(ws + off); off += (size_t)kD * kD * 2;
  __bf16* WoT = (__bf16*)(ws + off); off += (size_t)kD * kD * 2;
  __bf16* projbf = (__bf16*)(ws + off); off += (size_t)kM * kDH * 2;
  float* ctx  = (float*)(ws + off); off += (size_t)kB * kH * kM * kDH * 4;
  float* ksum = (float*)(ws + off); off += (size_t)kB * kH * kM * 4;
  float* sumpool = (float*)(ws + off); off += (size_t)kB * kD * 4;
  unsigned* maxenc = (unsigned*)(ws + off); off += (size_t)kB * kD * 4;

  k_pack<<<(4 * kD * kD + kM * kDH + 255) / 256, 256, 0, stream>>>(
      Wq, Wk, Wv, Wo, proj, WqT, WkT, WvT, WoT, projbf);
  k_ln<<<kRows / 8, 256, 0, stream>>>(x, g1, b1, xn);
  const int nz = kB * kH * kM * kDH + kB * kH * kM + kB * kD + kB * kD;
  k_zero<<<(nz + 255) / 256, 256, 0, stream>>>(ctx, nz); // ctx..maxenc contiguous
  k_qkv<<<dim3(kRows / 16, 3), 256, 0, stream>>>(
      xn, WqT, WkT, WvT, bq, bk, bv, mask, qb, kb, vb);
  k_ctx<<<kB * kH * 8, 256, 0, stream>>>(kb, vb, projbf, ctx, ksum);
  k_attn_out<<<dim3(kRows / 128, kH), 256, 0, stream>>>(qb, projbf, ctx, ksum, oa);
  k_wo_ln_pool<<<kRows / 16, 256, 0, stream>>>(
      oa, WoT, bo, x, g2, b2, mask, sumpool, maxenc);
  k_finalize<<<kB, 256, 0, stream>>>(mask, sumpool, maxenc, out);
}
